// Evolution_40200893890945
// MI455X (gfx1250) — matrix operand order repaired and checked by reading, not compiled, not run on hardware
//
#include <hip/hip_runtime.h>
#include <hip/hip_bf16.h>
#include <stdint.h>
#include <math.h>

// ---------------------------------------------------------------------------
// Deep-Snake contour evolution for MI455X (gfx1250, wave32, WMMA f16).
//   - circular convs  = 9 shifted GEMMs W_k(128x128) @ X(128xP)  [v_wmma]
//   - fusion/pred1/2  = K-chunked GEMMs (K up to 1024)           [v_wmma]
// Activations f16 in [n][p][c] layout -> B-fragments and C/D stores are
// contiguous LDS/global accesses; weights pre-swizzled -> contiguous A-frags.
// Weight taps / K-chunks are DOUBLE-BUFFERED via GLOBAL_LOAD_ASYNC_TO_LDS
// (ASYNCcnt + s_wait_asynccnt), overlapping L2 fetch with WMMA compute.
// ---------------------------------------------------------------------------

typedef _Float16 half_t;
typedef __attribute__((ext_vector_type(16))) _Float16 v16h;
typedef __attribute__((ext_vector_type(8)))  _Float16 v8h;
typedef __attribute__((ext_vector_type(8)))  float    v8f;
typedef int v4i128 __attribute__((vector_size(16)));  // builtin's b128 element

#define NPOLY 512
#define P0    40
#define PN    128
#define CCH   64
#define HH    128
#define WW    128

static const int DIL[7] = {1, 1, 1, 2, 2, 4, 4};

// ---------------------------------------------------------------------------
// Async global->LDS staging (CDNA5 async path, ASYNCcnt-tracked).
// Falls back to a synchronous float4 copy if the builtin is not declared.
// ---------------------------------------------------------------------------
#if defined(__gfx1250__) && __has_builtin(__builtin_amdgcn_global_load_async_to_lds_b128)
#define HAS_ASYNC_LDS 1
#else
#define HAS_ASYNC_LDS 0
#endif

__device__ __forceinline__ void stage_b128(half_t* lds_dst, const half_t* gsrc) {
#if HAS_ASYNC_LDS
  __builtin_amdgcn_global_load_async_to_lds_b128((v4i128*)gsrc, (v4i128*)lds_dst,
                                                 0, 0);
#else
  *(float4*)lds_dst = *(const float4*)gsrc;
#endif
}

__device__ __forceinline__ void stage_wait() {
#if HAS_ASYNC_LDS
#if __has_builtin(__builtin_amdgcn_s_wait_asynccnt)
  __builtin_amdgcn_s_wait_asynccnt(0);
#else
  asm volatile("s_wait_asynccnt 0" ::: "memory");
#endif
#endif
}

// ---------------------------------------------------------------------------
// Bilinear sampling (matches reference get_gcn_feature tap logic)
// ---------------------------------------------------------------------------
__device__ __forceinline__ float bilin(const float* __restrict__ F, int b, int ch,
                                       float px, float py) {
  const float* Fc = F + (((size_t)b * CCH + ch) * HH) * WW;
  float xs = px - 0.5f, ys = py - 0.5f;
  float fx = floorf(xs), fy = floorf(ys);
  int x0 = (int)fx, y0 = (int)fy;
  float wx = xs - fx, wy = ys - fy;
  float acc = 0.f;
#pragma unroll
  for (int dy = 0; dy < 2; ++dy)
#pragma unroll
    for (int dx = 0; dx < 2; ++dx) {
      int xi = x0 + dx, yi = y0 + dy;
      float w = (dx ? wx : 1.f - wx) * (dy ? wy : 1.f - wy);
      if (xi >= 0 && xi < WW && yi >= 0 && yi < HH)
        acc += w * Fc[yi * WW + xi];
    }
  return acc;
}

// ---------------------------------------------------------------------------
// Weight repack: f32 -> f16, A-fragment swizzle within each 32-wide K block.
// Lane order this code uses for A and B alike (contiguous k): element j, half hi -> K = j+16*hi.
// ---------------------------------------------------------------------------
__device__ __forceinline__ int swz32(int pos) {
  int hi = pos >> 4, j = pos & 15;
  return j + hi * 16;
}

// src [O][Cin][9] -> dst [9][O][Cpad] (swizzled, zero-padded)
__global__ void repack_circ(const float* __restrict__ src, half_t* __restrict__ dst,
                            int O, int Cin, int Cpad) {
  size_t total = (size_t)9 * O * Cpad;
  for (size_t i = (size_t)blockIdx.x * blockDim.x + threadIdx.x; i < total;
       i += (size_t)gridDim.x * blockDim.x) {
    int k = (int)(i / ((size_t)O * Cpad));
    int rem = (int)(i % ((size_t)O * Cpad));
    int o = rem / Cpad, pos = rem % Cpad;
    int c = (pos & ~31) + swz32(pos & 31);
    float v = (c < Cin) ? src[((size_t)o * Cin + c) * 9 + k] : 0.f;
    dst[i] = (half_t)v;
  }
}

// src [O][srcStride] (cols col0..col0+Cin) -> dst [O][Cin] swizzled
__global__ void repack_mat(const float* __restrict__ src, half_t* __restrict__ dst,
                           int O, int Cin, int srcStride, int col0) {
  size_t total = (size_t)O * Cin;
  for (size_t i = (size_t)blockIdx.x * blockDim.x + threadIdx.x; i < total;
       i += (size_t)gridDim.x * blockDim.x) {
    int o = (int)(i / Cin), pos = (int)(i % Cin);
    int c = (pos & ~31) + swz32(pos & 31);
    dst[i] = (half_t)src[(size_t)o * srcStride + col0 + c];
  }
}

// ---------------------------------------------------------------------------
// Circular dilated conv as WMMA:  Y[o,p] = relu(sum_k W_k @ X_shift) (+res)
// O = 128 fixed (8 waves = 8 M-tiles). X staged once; weight taps
// double-buffered with async-to-LDS so fetch of tap k+1 hides under tap k.
// ---------------------------------------------------------------------------
template <int CIN, int PT>
__global__ __launch_bounds__(256) void circ_conv_wmma(
    const half_t* __restrict__ Xg, size_t x_nstride, int x_rowstride, int x_coloff,
    half_t* __restrict__ Yg, size_t y_nstride, int y_rowstride, int y_coloff,
    const half_t* __restrict__ Wg,  // [9][128][CIN] swizzled
    const float* __restrict__ bias, int Pvalid, int dil, int residual) {
  constexpr int PRT = PT * 16;
  constexpr int WHALF = 128 * CIN;  // halves per weight tap
  extern __shared__ half_t lds[];
  half_t* xs = lds;             // [PRT][CIN]
  half_t* wb = lds + PRT * CIN; // 2 x [128][CIN] (double buffer)

  const int n = blockIdx.x;
  const int tid = threadIdx.x;
  const int wave = tid >> 5, lane = tid & 31;
  const int row = lane & 15, hi = lane >> 4;
  const int cw = CIN / 8;  // float4 per row

  // stage X tile (point-major, contiguous channels) - once per block
  const half_t* xb = Xg + (size_t)n * x_nstride + x_coloff;
  for (int i = tid; i < PRT * cw; i += 256) {
    int p = i / cw, j = i % cw;
    ((float4*)(xs + p * CIN))[j] = ((const float4*)(xb + (size_t)p * x_rowstride))[j];
  }

  // async-issue one full weight tap into an LDS buffer
  auto issueW = [&](int k, half_t* dst) {
    const half_t* wk = Wg + (size_t)k * WHALF;
    for (int i = tid; i < WHALF / 8; i += 256)
      stage_b128(dst + i * 8, wk + i * 8);
  };

  v8f acc[PT];
#pragma unroll
  for (int nt = 0; nt < PT; ++nt)
#pragma unroll
    for (int r = 0; r < 8; ++r) acc[nt][r] = 0.f;

  issueW(0, wb);
  stage_wait();
  __syncthreads();  // X + tap0 visible to all waves

  const int mtile = wave;  // O=128 -> 8 M-tiles
  for (int k = 0; k < 9; ++k) {
    half_t* ws = wb + (size_t)(k & 1) * WHALF;
    if (k < 8) issueW(k + 1, wb + (size_t)((k + 1) & 1) * WHALF);  // prefetch

    const int shift = (k - 4) * dil;
    int psv[PT];
#pragma unroll
    for (int nt = 0; nt < PT; ++nt) {
      int p = nt * 16 + row;
      psv[nt] = (p + shift + 4 * Pvalid) % Pvalid;  // circular wrap
    }
#pragma unroll
    for (int kk = 0; kk < CIN / 32; ++kk) {
      v16h a = *(const v16h*)(ws + (mtile * 16 + row) * CIN + kk * 32 + hi * 16);
#pragma unroll
      for (int nt = 0; nt < PT; ++nt) {
        v16h b = *(const v16h*)(xs + psv[nt] * CIN + kk * 32 + hi * 16);
        acc[nt] = __builtin_amdgcn_wmma_f32_16x16x32_f16(false, a, false, b,
                                                         (short)0, acc[nt],
                                                         false, false);
      }
    }
    stage_wait();     // tap k+1 landed (this wave's async ops)
    __syncthreads();  // all waves done reading tap k / writing tap k+1
  }

  // epilogue: bias + relu (+ residual), one b128 store per tile per lane
  const int cb = mtile * 16 + hi * 8;
  float4 b0 = *(const float4*)(bias + cb);
  float4 b1 = *(const float4*)(bias + cb + 4);
  float bv[8] = {b0.x, b0.y, b0.z, b0.w, b1.x, b1.y, b1.z, b1.w};
#pragma unroll
  for (int nt = 0; nt < PT; ++nt) {
    int p = nt * 16 + row;
    v8h res;
    if (residual) res = *(const v8h*)(xs + p * CIN + cb);
    v8h o8;
#pragma unroll
    for (int r = 0; r < 8; ++r) {
      float v = acc[nt][r] + bv[r];
      v = fmaxf(v, 0.f);
      if (residual) v += (float)res[r];
      o8[r] = (half_t)v;
    }
    *(v8h*)(Yg + (size_t)n * y_nstride + (size_t)p * y_rowstride + y_coloff + cb) = o8;
  }
}

// ---------------------------------------------------------------------------
// conv1x1 as K-chunked WMMA GEMM:  Y[p][o] = act(W(OxCIN) @ X[p][:] + bias)
// Both A and B K-chunks double-buffered via async-to-LDS.
// ---------------------------------------------------------------------------
template <int O, int CIN, int PT>
__global__ __launch_bounds__(256) void gemm_wmma(
    const half_t* __restrict__ Xg, size_t x_nstride, int x_rowstride,
    half_t* __restrict__ Yg, size_t y_nstride, int y_rowstride,
    const half_t* __restrict__ Wg,  // [O][CIN] swizzled
    const float* __restrict__ bias, int bias_per_n, int relu) {
  constexpr int KC = 64;
  constexpr int NC = CIN / KC;
  constexpr int PRT = PT * 16;
  constexpr int MT = O / 16;
  constexpr int T = MT * PT;
  constexpr int TMAX = (T + 7) / 8;
  extern __shared__ half_t lds[];
  half_t* xsb = lds;                 // 2 x [PRT][KC]
  half_t* wsb = lds + 2 * PRT * KC;  // 2 x [O][KC]

  const int n = blockIdx.x;
  const int tid = threadIdx.x;
  const int wave = tid >> 5, lane = tid & 31;
  const int row = lane & 15, hi = lane >> 4;
  const int cw = KC / 8;

  auto issue = [&](int ci, int buf) {
    half_t* xd = xsb + (size_t)buf * PRT * KC;
    half_t* wd = wsb + (size_t)buf * O * KC;
    const half_t* xsrc = Xg + (size_t)n * x_nstride + ci * KC;
    const half_t* wsrc = Wg + ci * KC;
    for (int i = tid; i < PRT * cw; i += 256) {
      int p = i / cw, j = i % cw;
      stage_b128(xd + p * KC + j * 8, xsrc + (size_t)p * x_rowstride + j * 8);
    }
    for (int i = tid; i < O * cw; i += 256) {
      int o = i / cw, j = i % cw;
      stage_b128(wd + o * KC + j * 8, wsrc + (size_t)o * CIN + j * 8);
    }
  };

  v8f acc[TMAX];
#pragma unroll
  for (int i = 0; i < TMAX; ++i)
#pragma unroll
    for (int r = 0; r < 8; ++r) acc[i][r] = 0.f;

  issue(0, 0);
  stage_wait();
  __syncthreads();

  for (int ci = 0; ci < NC; ++ci) {
    half_t* xs = xsb + (size_t)(ci & 1) * PRT * KC;
    half_t* ws = wsb + (size_t)(ci & 1) * O * KC;
    if (ci + 1 < NC) issue(ci + 1, (ci + 1) & 1);  // prefetch next chunk

#pragma unroll
    for (int kk = 0; kk < KC / 32; ++kk) {
#pragma unroll
      for (int i = 0; i < TMAX; ++i) {
        int t = wave + i * 8;
        if (t < T) {  // wave-uniform, EXEC stays all-ones
          int mt = t / PT, nt = t % PT;
          v16h a = *(const v16h*)(ws + (mt * 16 + row) * KC + kk * 32 + hi * 16);
          v16h b = *(const v16h*)(xs + (nt * 16 + row) * KC + kk * 32 + hi * 16);
          acc[i] = __builtin_amdgcn_wmma_f32_16x16x32_f16(false, a, false, b,
                                                          (short)0, acc[i],
                                                          false, false);
        }
      }
    }
    stage_wait();
    __syncthreads();
  }

#pragma unroll
  for (int i = 0; i < TMAX; ++i) {
    int t = wave + i * 8;
    if (t >= T) continue;
    int mt = t / PT, nt = t % PT;
    int cb = mt * 16 + hi * 8;
    const float* bp = bias_per_n ? (bias + (size_t)n * O + cb) : (bias + cb);
    float4 b0 = *(const float4*)(bp);
    float4 b1 = *(const float4*)(bp + 4);
    float bv[8] = {b0.x, b0.y, b0.z, b0.w, b1.x, b1.y, b1.z, b1.w};
    int p = nt * 16 + row;
    v8h o8;
#pragma unroll
    for (int r = 0; r < 8; ++r) {
      float v = acc[i][r] + bv[r];
      if (relu) v = fmaxf(v, 0.f);
      o8[r] = (half_t)v;
    }
    *(v8h*)(Yg + (size_t)n * y_nstride + (size_t)p * y_rowstride + cb) = o8;
  }
}

// ---------------------------------------------------------------------------
// Small VALU kernels
// ---------------------------------------------------------------------------
__global__ __launch_bounds__(128) void stats_k(const float* __restrict__ poly,
                                               float* __restrict__ stats, int P) {
  __shared__ float mnx[128], mny[128], mxx[128], mxy[128];
  int n = blockIdx.x, t = threadIdx.x;
  float x = 1e30f, y = 1e30f, X = -1e30f, Y = -1e30f;
  if (t < P) {
    x = poly[((size_t)n * P + t) * 2];
    y = poly[((size_t)n * P + t) * 2 + 1];
    X = x; Y = y;
  }
  mnx[t] = x; mny[t] = y; mxx[t] = X; mxy[t] = Y;
  __syncthreads();
  for (int s = 64; s > 0; s >>= 1) {
    if (t < s) {
      mnx[t] = fminf(mnx[t], mnx[t + s]);
      mny[t] = fminf(mny[t], mny[t + s]);
      mxx[t] = fmaxf(mxx[t], mxx[t + s]);
      mxy[t] = fmaxf(mxy[t], mxy[t + s]);
    }
    __syncthreads();
  }
  if (t == 0) {
    stats[n * 4 + 0] = mnx[0];
    stats[n * 4 + 1] = mny[0];
    stats[n * 4 + 2] = 0.5f * (mnx[0] + mxx[0]);
    stats[n * 4 + 3] = 0.5f * (mny[0] + mxy[0]);
  }
}

// init: x = concat(fuse(feat, center_feat), can_poly) padded to 96ch, f16
__global__ __launch_bounds__(128) void init_build_x(
    const float* __restrict__ cnn, const float* __restrict__ poly,
    const float* __restrict__ stats, const int* __restrict__ ind,
    const float* __restrict__ fuse_w, const float* __restrict__ fuse_b,
    half_t* __restrict__ X96) {
  __shared__ float sf[128];
  int n = blockIdx.x, p = blockIdx.y, t = threadIdx.x;
  int b = ind[n];
  float px = poly[((size_t)n * P0 + p) * 2];
  float py = poly[((size_t)n * P0 + p) * 2 + 1];
  if (t < 64) sf[t] = bilin(cnn, b, t, px, py);
  else        sf[t] = bilin(cnn, b, t - 64, stats[n * 4 + 2], stats[n * 4 + 3]);
  __syncthreads();
  half_t* xr = X96 + ((size_t)n * PN + p) * 96;
  if (t < 64) {
    const float* w = fuse_w + t * 128;
    float acc = fuse_b[t];
    for (int c = 0; c < 128; ++c) acc += w[c] * sf[c];
    xr[t] = (half_t)acc;
  } else if (t == 64) xr[64] = (half_t)(px - stats[n * 4 + 0]);
  else if (t == 65)   xr[65] = (half_t)(py - stats[n * 4 + 1]);
  else if (t < 96)    xr[t] = (half_t)0.f;
}

// evolve: x = concat(feat, (poly - min) * RO) padded to 96ch
__global__ __launch_bounds__(96) void evolve_build_x(
    const float* __restrict__ cnn, const float* __restrict__ poly,
    const float* __restrict__ stats, const int* __restrict__ ind,
    half_t* __restrict__ X96, float ro) {
  int n = blockIdx.x, p = blockIdx.y, t = threadIdx.x;
  float px = poly[((size_t)n * PN + p) * 2];
  float py = poly[((size_t)n * PN + p) * 2 + 1];
  half_t* xr = X96 + ((size_t)n * PN + p) * 96;
  if (t < 64)       xr[t] = (half_t)bilin(cnn, ind[n], t, px, py);
  else if (t == 64) xr[64] = (half_t)((px - stats[n * 4 + 0]) * ro);
  else if (t == 65) xr[65] = (half_t)((py - stats[n * 4 + 1]) * ro);
  else              xr[t] = (half_t)0.f;
}

__global__ __launch_bounds__(256) void fus_max(const half_t* __restrict__ FUS,
                                               float* __restrict__ GLOB, int Pvalid) {
  int n = blockIdx.x, o = threadIdx.x;
  float m = -1e30f;
  for (int p = 0; p < Pvalid; ++p)
    m = fmaxf(m, (float)FUS[((size_t)n * PN + p) * 256 + o]);
  GLOB[n * 256 + o] = m;
}

// pbias[n][o] = b1[o] + W1[o][0:256] . glob[n]   (global-feature GEMV)
__global__ __launch_bounds__(256) void pbias_k(const float* __restrict__ W1,
                                               const float* __restrict__ b1,
                                               const float* __restrict__ GLOB,
                                               float* __restrict__ PB) {
  int n = blockIdx.x, o = threadIdx.x;
  const float* g = GLOB + (size_t)n * 256;
  const float* w = W1 + (size_t)o * 1280;
  float acc = b1[o];
  for (int c = 0; c < 256; ++c) acc += w[c] * g[c];
  PB[(size_t)n * 256 + o] = acc;
}

// pred3 (2x64) + polygon update:  out = alpha*ipoly + W3 @ h2 + b3
__global__ __launch_bounds__(64) void pred3_update(
    const half_t* __restrict__ H2, const float* __restrict__ w3,
    const float* __restrict__ b3, const float* __restrict__ ipoly, float alpha,
    float* __restrict__ opoly, int P) {
  __shared__ float s0[64], s1[64];
  int n = blockIdx.x, p = blockIdx.y, t = threadIdx.x;
  float v = (float)H2[((size_t)n * PN + p) * 64 + t];
  s0[t] = v * w3[t];
  s1[t] = v * w3[64 + t];
  __syncthreads();
  for (int s = 32; s > 0; s >>= 1) {
    if (t < s) { s0[t] += s0[t + s]; s1[t] += s1[t + s]; }
    __syncthreads();
  }
  if (t == 0) {
    size_t base = ((size_t)n * P + p) * 2;
    opoly[base]     = alpha * ipoly[base]     + s0[0] + b3[0];
    opoly[base + 1] = alpha * ipoly[base + 1] + s1[0] + b3[1];
  }
}

// uniform_resample 40 -> 128 points per polygon
__global__ __launch_bounds__(128) void resample_k(const float* __restrict__ in,
                                                  float* __restrict__ out) {
  __shared__ float px[P0], py[P0], cl[P0 + 1];
  int n = blockIdx.x, t = threadIdx.x;
  if (t < P0) {
    px[t] = in[((size_t)n * P0 + t) * 2];
    py[t] = in[((size_t)n * P0 + t) * 2 + 1];
  }
  __syncthreads();
  if (t == 0) {
    cl[0] = 0.f;
    for (int i = 0; i < P0; ++i) {
      int j = (i + 1) % P0;
      float dx = px[j] - px[i], dy = py[j] - py[i];
      cl[i + 1] = cl[i] + sqrtf(dx * dx + dy * dy);
    }
  }
  __syncthreads();
  float tt = cl[P0] * ((float)t / (float)PN);
  int idx = 0;
  for (int i = 1; i <= P0; ++i)
    if (cl[i] <= tt) idx = i;  // searchsorted(right) - 1
  if (idx > P0 - 1) idx = P0 - 1;
  float el = cl[idx + 1] - cl[idx];
  float frac = (tt - cl[idx]) / fmaxf(el, 1e-6f);
  int j = (idx + 1) % P0;
  out[((size_t)n * PN + t) * 2]     = px[idx] + frac * (px[j] - px[idx]);
  out[((size_t)n * PN + t) * 2 + 1] = py[idx] + frac * (py[j] - py[idx]);
}

__global__ void scale_k(const float* __restrict__ in, float* __restrict__ out,
                        float s, int n) {
  int i = blockIdx.x * blockDim.x + threadIdx.x;
  if (i < n) out[i] = in[i] * s;
}

// ---------------------------------------------------------------------------
// Host side
// ---------------------------------------------------------------------------
struct SnakeIn {
  const float *fusion_b, *fusion_w, *head_b, *head_w;
  const float *pred_b1, *pred_b2, *pred_b3, *pred_w1, *pred_w2, *pred_w3;
  const float *res_b[7], *res_w[7];
};
static SnakeIn snake_in(void* const* d_in, int base) {
  SnakeIn s;
  s.fusion_b = (const float*)d_in[base + 0];
  s.fusion_w = (const float*)d_in[base + 1];
  s.head_b   = (const float*)d_in[base + 2];
  s.head_w   = (const float*)d_in[base + 3];
  s.pred_b1  = (const float*)d_in[base + 4];
  s.pred_b2  = (const float*)d_in[base + 5];
  s.pred_b3  = (const float*)d_in[base + 6];
  s.pred_w1  = (const float*)d_in[base + 7];
  s.pred_w2  = (const float*)d_in[base + 8];
  s.pred_w3  = (const float*)d_in[base + 9];
  for (int i = 0; i < 7; ++i) s.res_b[i] = (const float*)d_in[base + 10 + i];
  for (int i = 0; i < 7; ++i) s.res_w[i] = (const float*)d_in[base + 17 + i];
  return s;
}

struct SnakeW { half_t *headW, *resW[7], *fusW, *p1W, *p2W; };
struct Bufs { half_t *X96, *STATE, *FUS, *H2; float *GLOB, *PBIAS; };

template <int PT>
static void run_snake_t(const SnakeIn& si, const SnakeW& sw, const Bufs& B,
                        int Pvalid, hipStream_t s) {
  constexpr int PRT = PT * 16;
  circ_conv_wmma<96, PT><<<NPOLY, 256, (PRT * 96 + 2 * 128 * 96) * 2, s>>>(
      B.X96, (size_t)PN * 96, 96, 0, B.STATE, (size_t)PN * 1024, 1024, 0,
      sw.headW, si.head_b, Pvalid, 1, 0);
  for (int i = 0; i < 7; ++i)
    circ_conv_wmma<128, PT><<<NPOLY, 256, (PRT * 128 + 2 * 128 * 128) * 2, s>>>(
        B.STATE, (size_t)PN * 1024, 1024, 128 * i, B.STATE, (size_t)PN * 1024,
        1024, 128 * (i + 1), sw.resW[i], si.res_b[i], Pvalid, DIL[i], 1);
  gemm_wmma<256, 1024, PT><<<NPOLY, 256, 2 * (PRT + 256) * 64 * 2, s>>>(
      B.STATE, (size_t)PN * 1024, 1024, B.FUS, (size_t)PN * 256, 256, sw.fusW,
      si.fusion_b, 0, 0);
  fus_max<<<NPOLY, 256, 0, s>>>(B.FUS, B.GLOB, Pvalid);
  pbias_k<<<NPOLY, 256, 0, s>>>(si.pred_w1, si.pred_b1, B.GLOB, B.PBIAS);
  gemm_wmma<256, 1024, PT><<<NPOLY, 256, 2 * (PRT + 256) * 64 * 2, s>>>(
      B.STATE, (size_t)PN * 1024, 1024, B.FUS /* reuse as H1 */,
      (size_t)PN * 256, 256, sw.p1W, B.PBIAS, 1, 1);
  gemm_wmma<64, 256, PT><<<NPOLY, 256, 2 * (PRT + 64) * 64 * 2, s>>>(
      B.FUS, (size_t)PN * 256, 256, B.H2, (size_t)PN * 64, 64, sw.p2W,
      si.pred_b2, 0, 1);
}

extern "C" void kernel_launch(void* const* d_in, const int* in_sizes, int n_in,
                              void* d_out, int out_size, void* d_ws, size_t ws_size,
                              hipStream_t stream) {
  (void)in_sizes; (void)n_in; (void)out_size; (void)ws_size;
  const float* cnn  = (const float*)d_in[0];
  const float* i_it = (const float*)d_in[1];
  const int*   ind  = (const int*)d_in[3];
  // params flattened (sorted keys): evolve_gcn, evolve_gcn0, fuse_b, fuse_w,
  // init_gcn0, init_gcn1, init_gcn2 (each snake = 24 leaves, sorted keys)
  SnakeIn S[5] = {snake_in(d_in, 54), snake_in(d_in, 78), snake_in(d_in, 102),
                  snake_in(d_in, 4),  snake_in(d_in, 28)};
  const float* fuse_b = (const float*)d_in[52];
  const float* fuse_w = (const float*)d_in[53];

  // -------- workspace carve --------
  uint8_t* wsb = (uint8_t*)d_ws;
  size_t off = 0;
  auto carve = [&](size_t bytes) -> void* {
    void* p = wsb + off;
    off = (off + bytes + 255) & ~(size_t)255;
    return p;
  };
  SnakeW W[5];
  for (int s = 0; s < 5; ++s) {
    W[s].headW = (half_t*)carve((size_t)9 * 128 * 96 * 2);
    for (int i = 0; i < 7; ++i) W[s].resW[i] = (half_t*)carve((size_t)9 * 128 * 128 * 2);
    W[s].fusW = (half_t*)carve((size_t)256 * 1024 * 2);
    W[s].p1W  = (half_t*)carve((size_t)256 * 1024 * 2);
    W[s].p2W  = (half_t*)carve((size_t)64 * 256 * 2);
  }
  Bufs B;
  B.X96   = (half_t*)carve((size_t)NPOLY * PN * 96 * 2);
  B.STATE = (half_t*)carve((size_t)NPOLY * PN * 1024 * 2);
  B.FUS   = (half_t*)carve((size_t)NPOLY * PN * 256 * 2);  // also H1
  B.H2    = (half_t*)carve((size_t)NPOLY * PN * 64 * 2);
  B.GLOB  = (float*)carve((size_t)NPOLY * 256 * 4);
  B.PBIAS = (float*)carve((size_t)NPOLY * 256 * 4);
  float* polyA = (float*)carve((size_t)NPOLY * P0 * 2 * 4);
  float* polyB = (float*)carve((size_t)NPOLY * P0 * 2 * 4);
  float* STATS = (float*)carve((size_t)NPOLY * 4 * 4);
  float* P128A = (float*)carve((size_t)NPOLY * PN * 2 * 4);
  float* PYH   = (float*)carve((size_t)NPOLY * PN * 2 * 4);
  float* PY0   = (float*)carve((size_t)NPOLY * PN * 2 * 4);
  float* PY1   = (float*)carve((size_t)NPOLY * PN * 2 * 4);

  // -------- weight repack (f32 -> f16, swizzled) --------
  for (int s = 0; s < 5; ++s) {
    repack_circ<<<1024, 256, 0, stream>>>(S[s].head_w, W[s].headW, 128, 66, 96);
    for (int i = 0; i < 7; ++i)
      repack_circ<<<1024, 256, 0, stream>>>(S[s].res_w[i], W[s].resW[i], 128, 128, 128);
    repack_mat<<<1024, 256, 0, stream>>>(S[s].fusion_w, W[s].fusW, 256, 1024, 1024, 0);
    repack_mat<<<1024, 256, 0, stream>>>(S[s].pred_w1, W[s].p1W, 256, 1024, 1280, 256);
    repack_mat<<<256, 256, 0, stream>>>(S[s].pred_w2, W[s].p2W, 64, 256, 256, 0);
  }

  // -------- 3 init steps on 40-point polygons --------
  (void)hipMemcpyAsync(polyA, i_it, (size_t)NPOLY * P0 * 2 * 4,
                       hipMemcpyDeviceToDevice, stream);
  for (int step = 0; step < 3; ++step) {
    stats_k<<<NPOLY, 128, 0, stream>>>(polyA, STATS, P0);
    init_build_x<<<dim3(NPOLY, P0), 128, 0, stream>>>(cnn, polyA, STATS, ind,
                                                      fuse_w, fuse_b, B.X96);
    run_snake_t<3>(S[step], W[step], B, P0, stream);
    pred3_update<<<dim3(NPOLY, P0), 64, 0, stream>>>(
        B.H2, S[step].pred_w3, S[step].pred_b3, polyA, 1.0f, polyB, P0);
    float* t = polyA; polyA = polyB; polyB = t;
  }
  // polyA == ex_pred

  // -------- resample to 128 points --------
  resample_k<<<NPOLY, 128, 0, stream>>>(polyA, P128A);

  // -------- evolve 1 (evolve_gcn) --------
  stats_k<<<NPOLY, 128, 0, stream>>>(P128A, STATS, PN);
  evolve_build_x<<<dim3(NPOLY, PN), 96, 0, stream>>>(cnn, P128A, STATS, ind,
                                                     B.X96, 4.0f);
  run_snake_t<8>(S[3], W[3], B, PN, stream);
  pred3_update<<<dim3(NPOLY, PN), 64, 0, stream>>>(
      B.H2, S[3].pred_w3, S[3].pred_b3, P128A, 4.0f, PY0, PN);

  // -------- evolve 2 (evolve_gcn0) on py/RO --------
  scale_k<<<(NPOLY * PN * 2 + 255) / 256, 256, 0, stream>>>(PY0, PYH, 0.25f,
                                                            NPOLY * PN * 2);
  stats_k<<<NPOLY, 128, 0, stream>>>(PYH, STATS, PN);
  evolve_build_x<<<dim3(NPOLY, PN), 96, 0, stream>>>(cnn, PYH, STATS, ind,
                                                     B.X96, 4.0f);
  run_snake_t<8>(S[4], W[4], B, PN, stream);
  pred3_update<<<dim3(NPOLY, PN), 64, 0, stream>>>(
      B.H2, S[4].pred_w3, S[4].pred_b3, PYH, 4.0f, PY1, PN);

  // -------- pack outputs: ex_pred | py0 | py1 --------
  float* out = (float*)d_out;
  (void)hipMemcpyAsync(out, polyA, (size_t)NPOLY * P0 * 2 * 4,
                       hipMemcpyDeviceToDevice, stream);
  (void)hipMemcpyAsync(out + NPOLY * P0 * 2, PY0, (size_t)NPOLY * PN * 2 * 4,
                       hipMemcpyDeviceToDevice, stream);
  (void)hipMemcpyAsync(out + NPOLY * P0 * 2 + NPOLY * PN * 2, PY1,
                       (size_t)NPOLY * PN * 2 * 4, hipMemcpyDeviceToDevice,
                       stream);
}
